// GCC_54786602828345
// MI455X (gfx1250) — matrix operand order repaired and checked by reading, not compiled, not run on hardware
//
#include <hip/hip_runtime.h>
#include <math.h>

// ---- problem constants (shapes fixed by the reference) ----
#define D_IN   32
#define D_H    64
#define D_OUT  32
#define LD0    68      // h0 stored with 65 real cols padded to 68
#define KP0    96      // layer-0 GEMM1 K padded to 96 (3 chunks of 32)
#define MAXDEG 128
#define N_GIN  5

typedef __attribute__((ext_vector_type(16))) __bf16 v16bf;
typedef __attribute__((ext_vector_type(8)))  __bf16 v8bf;
typedef __attribute__((ext_vector_type(8)))  float  v8f;

// ------------------------------------------------------------------
// utility kernels
// ------------------------------------------------------------------
__global__ void k_zero(float* __restrict__ p, size_t n) {
  size_t i = (size_t)blockIdx.x * blockDim.x + threadIdx.x;
  if (i < n) p[i] = 0.0f;
}

__global__ void k_degree(const int* __restrict__ src, int* __restrict__ deg, int E) {
  int e = blockIdx.x * blockDim.x + threadIdx.x;
  if (e < E) atomicAdd(&deg[src[e]], 1);
}

__global__ void k_build_h0(const float* __restrict__ x, const int* __restrict__ deg,
                           const float* __restrict__ deg_table, float* __restrict__ h0, int N) {
  int i = blockIdx.x * blockDim.x + threadIdx.x;
  if (i >= N) return;
  int d = deg[i]; d = d > MAXDEG ? MAXDEG : d;
  float* r = h0 + (size_t)i * LD0;
#pragma unroll
  for (int c = 0; c < D_IN; ++c) r[c] = x[(size_t)i * D_IN + c];
#pragma unroll
  for (int c = 0; c < 32; ++c) r[D_IN + c] = deg_table[d * 32 + c];
  r[64] = 0.f; r[65] = 0.f; r[66] = 0.f; r[67] = 0.f;
}

__global__ void k_ego(const int* __restrict__ root, float* __restrict__ h0, int G) {
  int g = blockIdx.x * blockDim.x + threadIdx.x;
  if (g < G) h0[(size_t)root[g] * LD0 + 64] = 1.0f;   // idempotent set
}

__global__ void k_pool(const float* __restrict__ h, int ld, int cols,
                       const int* __restrict__ batch, float* __restrict__ pooled,
                       int pld, int N) {
  unsigned idx = blockIdx.x * blockDim.x + threadIdx.x;
  unsigned tot = (unsigned)N * (unsigned)cols;
  if (idx >= tot) return;
  unsigned row = idx / (unsigned)cols;
  unsigned c   = idx - row * (unsigned)cols;
  atomicAdd(&pooled[(size_t)batch[row] * pld + c], h[(size_t)row * ld + c]);
}

// agg[dst] += h[src]   (f32 L2 atomics; HBM-bound, this is the hot loop)
__global__ void k_edge_agg(const int* __restrict__ src, const int* __restrict__ dst,
                           const float* __restrict__ h, float* __restrict__ agg,
                           unsigned ld, unsigned total) {
  unsigned idx = blockIdx.x * blockDim.x + threadIdx.x;
  if (idx >= total) return;
  unsigned e = idx / ld;
  unsigned c = idx - e * ld;
  atomicAdd(&agg[(size_t)dst[e] * ld + c], h[(size_t)src[e] * ld + c]);
}

// ------------------------------------------------------------------
// Fused WMMA GEMM:  out[N,64] = act(A) @ W[K,64] + bias
//   act(A) = (A1 + A2)      when useBN == 0   (GIN aggregate)
//   act(A) = relu(bn(A1))   when useBN == 1   (stats from bnSum/bnSq)
// Also emits per-feature sum / sum-of-squares of `out` for the next BN.
// 128 threads = 4 waves; each wave computes a 16(M) x 64(N) strip with
// v_wmma_f32_16x16x32_bf16 (4 N-tiles x KP/32 K-chunks).
// W is pre-packed in LDS in per-lane fragment order so each B fragment is
// one contiguous 32-byte v16bf load (2x ds_load_b128, no scalar u16 loads).
// ------------------------------------------------------------------
__global__ __launch_bounds__(128)
void k_gin_gemm(const float* __restrict__ A1, const float* __restrict__ A2,
                int ldA, int Kcols, int KP,
                const float* __restrict__ W, int Kw,
                const float* __restrict__ bias,
                const float* __restrict__ bnSum, const float* __restrict__ bnSq,
                const float* __restrict__ gamma, const float* __restrict__ beta,
                int useBN, float invN,
                float* __restrict__ out,
                float* __restrict__ outSum, float* __restrict__ outSq,
                int N)
{
  // packed W fragments: [(chunk*4 + tile)*32 + lane][16] bf16
  __shared__ __bf16 ldsWp[3 * 4 * 32 * 16];       // 12 KB
  __shared__ __bf16 ldsA[4][16 * KP0];            // 12 KB (per-wave A tiles)
  __shared__ float  ldsScale[D_H], ldsShift[D_H];
  __shared__ float  ldsSum[D_H], ldsSq[D_H];

  const int tid  = threadIdx.x;
  const int lane = tid & 31;
  const int wave = tid >> 5;
  const int half = lane >> 4;   // 0: lanes 0-15, 1: lanes 16-31
  const int m    = lane & 15;

  // ---- stage W fragments (bf16), zero beyond Kw rows ----
  // ldsWp[((c*4+t)*32+L)*16 + e] = W[32c + 8*(L>>4) + (e&7) + 16*(e>>3)][16t + (L&15)]
  const int nCh = KP >> 5;
  for (int p = tid; p < (nCh << 11); p += 128) {
    int e = p & 15;
    int L = (p >> 4) & 31;
    int t = (p >> 9) & 3;
    int c = p >> 11;
    int k = (c << 5) + ((L >> 4) << 3) + (e & 7) + ((e >> 3) << 4);
    int n = (t << 4) + (L & 15);
    float w = (k < Kw) ? W[k * D_H + n] : 0.0f;
    ldsWp[p] = (__bf16)w;
  }
  if (useBN && tid < D_H) {
    float mean = bnSum[tid] * invN;
    float var  = bnSq[tid] * invN - mean * mean;
    float sc   = gamma[tid] * rsqrtf(var + 1e-5f);
    ldsScale[tid] = sc;
    ldsShift[tid] = beta[tid] - mean * sc;
  }
  if (tid < D_H) { ldsSum[tid] = 0.f; ldsSq[tid] = 0.f; }
  __syncthreads();

  // ---- stage this wave's A tile (16 rows), each lane fills half a row ----
  const int rowBase = (blockIdx.x * 4 + wave) * 16;
  __bf16* Aw = &ldsA[wave][0];
  const int cHalf = KP >> 1;                  // 48 or 32 (multiple of 8)
  const int cBeg  = half * cHalf;
  const int gRowM = rowBase + m;
  for (int c0 = cBeg; c0 < cBeg + cHalf; c0 += 8) {
    v8bf pk;
#pragma unroll
    for (int j = 0; j < 8; ++j) {
      int c = c0 + j;
      float v = 0.0f;
      if (gRowM < N && c < Kcols) {
        float a = A1[(size_t)gRowM * ldA + c];
        if (A2) a += A2[(size_t)gRowM * ldA + c];
        if (useBN) { float t = a * ldsScale[c] + ldsShift[c]; a = t > 0.f ? t : 0.f; }
        v = a;
      }
      pk[j] = (__bf16)v;
    }
    *(v8bf*)(Aw + m * KP + c0) = pk;          // 16B aligned
  }
  __syncthreads();

  // ---- WMMA accumulation: 4 output N-tiles per wave ----
  v8f acc[4] = {v8f{}, v8f{}, v8f{}, v8f{}};
  for (int kc = 0; kc < nCh; ++kc) {
    // A fragment = two contiguous 16B runs (ISA 7.12.2 16-bit A 16x32 layout)
    const __bf16* arow = Aw + m * KP + (kc << 5) + (half << 3);
    v8bf alo = *(const v8bf*)(arow);
    v8bf ahi = *(const v8bf*)(arow + 16);
    v16bf a = __builtin_shufflevector(alo, ahi,
                                      0, 1, 2, 3, 4, 5, 6, 7,
                                      8, 9, 10, 11, 12, 13, 14, 15);
#pragma unroll
    for (int t = 0; t < 4; ++t) {
      // B fragment: one contiguous 32B v16bf load from packed W
      const v16bf* pb =
          (const v16bf*)(ldsWp + (((((kc << 2) + t) << 5) | lane) << 4));
      v16bf b = *pb;
      acc[t] = __builtin_amdgcn_wmma_f32_16x16x32_bf16(
          /*neg_a=*/false, a, /*neg_b=*/false, b,
          /*c_mod=*/(short)0, acc[t], /*reuse_a=*/false, /*reuse_b=*/false);
    }
  }

  // ---- store D + bias; accumulate BN stats (LDS reduce, 1 atomic/feature) ----
  float sT[4], qT[4];
#pragma unroll
  for (int t = 0; t < 4; ++t) {
    int n = (t << 4) + m;
    float bv = bias[n];
    float s = 0.f, s2 = 0.f;
#pragma unroll
    for (int v = 0; v < 8; ++v) {
      int row = rowBase + v + half * 8;       // C layout: VGPR v -> M = v + 8*half
      if (row < N) {
        float val = acc[t][v] + bv;
        out[(size_t)row * D_H + n] = val;
        s += val; s2 += val * val;
      }
    }
    sT[t] = s; qT[t] = s2;
  }
  if (outSum) {
#pragma unroll
    for (int t = 0; t < 4; ++t) {
      int n = (t << 4) + m;
      atomicAdd(&ldsSum[n], sT[t]);
      atomicAdd(&ldsSq[n], qT[t]);
    }
    __syncthreads();
    if (tid < D_H) {
      atomicAdd(&outSum[tid], ldsSum[tid]);
      atomicAdd(&outSq[tid], ldsSq[tid]);
    }
  }
}

// ------------------------------------------------------------------
// Elementwise BN + ReLU; optional output stats; optional graph pooling
// (batch is sorted -> run-length compress pooling atomics).
// Block = 256 threads handles 64 nodes x 64 features.
// ------------------------------------------------------------------
__global__ __launch_bounds__(256)
void k_bn_relu(const float* __restrict__ in,
               const float* __restrict__ bnSum, const float* __restrict__ bnSq,
               const float* __restrict__ gamma, const float* __restrict__ beta,
               float invN, float* __restrict__ out,
               float* __restrict__ outSum, float* __restrict__ outSq,
               float* __restrict__ pooled, const int* __restrict__ batch, int N)
{
  __shared__ float sS[256], sQ[256];
  const int f    = threadIdx.x & 63;
  const int part = threadIdx.x >> 6;          // 0..3
  const int base = blockIdx.x * 64;

  float mean = bnSum[f] * invN;
  float var  = bnSq[f] * invN - mean * mean;
  float sc   = gamma[f] * rsqrtf(var + 1e-5f);
  float sh   = beta[f] - mean * sc;

  float s = 0.f, s2 = 0.f, pAcc = 0.f;
  int curG = -1;
  for (int j = 0; j < 16; ++j) {
    int row = base + part * 16 + j;
    if (row >= N) break;
    float v = in[(size_t)row * D_H + f] * sc + sh;
    v = v > 0.f ? v : 0.f;
    out[(size_t)row * D_H + f] = v;
    s += v; s2 += v * v;
    if (pooled) {
      int g = batch[row];
      if (g != curG) {
        if (curG >= 0) atomicAdd(&pooled[(size_t)curG * D_H + f], pAcc);
        curG = g; pAcc = 0.f;
      }
      pAcc += v;
    }
  }
  if (pooled && curG >= 0) atomicAdd(&pooled[(size_t)curG * D_H + f], pAcc);

  if (outSum) {
    sS[threadIdx.x] = s; sQ[threadIdx.x] = s2;
    __syncthreads();
    if (part == 0) {
      float ts = sS[f] + sS[64 + f] + sS[128 + f] + sS[192 + f];
      float tq = sQ[f] + sQ[64 + f] + sQ[128 + f] + sQ[192 + f];
      atomicAdd(&outSum[f], ts);
      atomicAdd(&outSq[f], tq);
    }
  }
}

// ------------------------------------------------------------------
// Prediction heads + L2 normalize. One thread per graph.
// ------------------------------------------------------------------
__global__ __launch_bounds__(128)
void k_score(const float* __restrict__ pooled0, const float* __restrict__ pooledR,
             const float* __restrict__ pw0, const float* __restrict__ pb0,
             const float* __restrict__ pwr, const float* __restrict__ pbr,
             float* __restrict__ outp, int G)
{
  int g = blockIdx.x * blockDim.x + threadIdx.x;
  if (g >= G) return;
  float acc[D_OUT];
#pragma unroll
  for (int o = 0; o < D_OUT; ++o) acc[o] = pb0[o];
  for (int l = 0; l < N_GIN; ++l)
#pragma unroll
    for (int o = 0; o < D_OUT; ++o) acc[o] += pbr[l * D_OUT + o];

  for (int k = 0; k < 65; ++k) {
    float p = pooled0[(size_t)g * LD0 + k];
#pragma unroll
    for (int o = 0; o < D_OUT; ++o) acc[o] += p * pw0[k * D_OUT + o];
  }
  for (int l = 0; l < N_GIN; ++l) {
    const float* pl = pooledR + (size_t)l * G * D_H;
    const float* wl = pwr + (size_t)l * D_H * D_OUT;
    for (int k = 0; k < D_H; ++k) {
      float p = pl[(size_t)g * D_H + k];
#pragma unroll
      for (int o = 0; o < D_OUT; ++o) acc[o] += p * wl[k * D_OUT + o];
    }
  }
  float nn = 0.f;
#pragma unroll
  for (int o = 0; o < D_OUT; ++o) nn += acc[o] * acc[o];
  float inv = 1.0f / fmaxf(sqrtf(nn), 1e-5f);
#pragma unroll
  for (int o = 0; o < D_OUT; ++o) outp[(size_t)g * D_OUT + o] = acc[o] * inv;
}

// ------------------------------------------------------------------
static inline int cdiv_i(long long a, long long b) { return (int)((a + b - 1) / b); }

extern "C" void kernel_launch(void* const* d_in, const int* in_sizes, int n_in,
                              void* d_out, int out_size, void* d_ws, size_t ws_size,
                              hipStream_t stream)
{
  const float* x         = (const float*)d_in[0];
  const int*   eidx      = (const int*)  d_in[1];
  const int*   batch     = (const int*)  d_in[2];
  const int*   root      = (const int*)  d_in[3];
  const float* deg_table = (const float*)d_in[4];
  const float* w1_0      = (const float*)d_in[5];
  const float* b1_0      = (const float*)d_in[6];
  const float* w2_0      = (const float*)d_in[7];
  const float* b2_0      = (const float*)d_in[8];
  const float* w1_r      = (const float*)d_in[9];
  const float* b1_r      = (const float*)d_in[10];
  const float* w2_r      = (const float*)d_in[11];
  const float* b2_r      = (const float*)d_in[12];
  const float* bn_g      = (const float*)d_in[13];
  const float* bn_b      = (const float*)d_in[14];
  const float* pw0       = (const float*)d_in[15];
  const float* pb0       = (const float*)d_in[16];
  const float* pwr       = (const float*)d_in[17];
  const float* pbr       = (const float*)d_in[18];

  const int N = in_sizes[0] / D_IN;
  const int E = in_sizes[1] / 2;
  const int G = in_sizes[3];
  const int* src = eidx;
  const int* dst = eidx + E;
  const float invN = 1.0f / (float)N;

  // ---- carve workspace ----
  char* wsp = (char*)d_ws;
  auto carve = [&](size_t bytes) -> void* {
    void* p = wsp;
    wsp += (bytes + 255) & ~(size_t)255;
    return p;
  };
  int*   deg     = (int*)  carve((size_t)N * 4);
  float* h0      = (float*)carve((size_t)N * LD0 * 4);
  float* agg     = (float*)carve((size_t)N * LD0 * 4);
  float* t1      = (float*)carve((size_t)N * D_H * 4);
  float* t2      = (float*)carve((size_t)N * D_H * 4);
  float* hc      = (float*)carve((size_t)N * D_H * 4);
  float* stats   = (float*)carve(6 * 64 * 4);
  float* pooled0 = (float*)carve((size_t)G * LD0 * 4);
  float* pooledR = (float*)carve((size_t)N_GIN * G * D_H * 4);

  // ---- degrees + h0 ----
  k_zero<<<cdiv_i(N, 256), 256, 0, stream>>>((float*)deg, (size_t)N);
  k_degree<<<cdiv_i(E, 256), 256, 0, stream>>>(src, deg, E);
  k_build_h0<<<cdiv_i(N, 256), 256, 0, stream>>>(x, deg, deg_table, h0, N);
  k_ego<<<cdiv_i(G, 128), 128, 0, stream>>>(root, h0, G);

  // ---- pooled buffers ----
  k_zero<<<cdiv_i((long long)G * LD0, 256), 256, 0, stream>>>(pooled0, (size_t)G * LD0);
  k_zero<<<cdiv_i((long long)N_GIN * G * D_H, 256), 256, 0, stream>>>(pooledR, (size_t)N_GIN * G * D_H);
  k_pool<<<cdiv_i((long long)N * 65, 256), 256, 0, stream>>>(h0, LD0, 65, batch, pooled0, LD0, N);

  const int gemmBlocks = cdiv_i(N, 64);
  const int bnBlocks   = cdiv_i(N, 64);

  for (int l = 0; l < N_GIN; ++l) {
    const float* hin = (l == 0) ? h0 : hc;
    const int ldin = (l == 0) ? LD0 : D_H;
    const int KP   = (l == 0) ? KP0 : D_H;
    const int Kw1  = (l == 0) ? 65 : D_H;
    const float* w1 = (l == 0) ? w1_0 : (w1_r + (size_t)(l - 1) * D_H * D_H);
    const float* b1 = (l == 0) ? b1_0 : (b1_r + (size_t)(l - 1) * D_H);
    const float* w2 = (l == 0) ? w2_0 : (w2_r + (size_t)(l - 1) * D_H * D_H);
    const float* b2 = (l == 0) ? b2_0 : (b2_r + (size_t)(l - 1) * D_H);
    const float* gl = bn_g + (size_t)l * 3 * D_H;
    const float* bl = bn_b + (size_t)l * 3 * D_H;

    // zero agg + per-layer BN stat accumulators
    k_zero<<<cdiv_i((long long)N * ldin, 256), 256, 0, stream>>>(agg, (size_t)N * ldin);
    k_zero<<<2, 256, 0, stream>>>(stats, (size_t)6 * 64);

    // edge scatter-sum (HBM-bound hot loop)
    unsigned etot = (unsigned)E * (unsigned)ldin;
    k_edge_agg<<<cdiv_i((long long)etot, 256), 256, 0, stream>>>(src, dst, hin, agg,
                                                                 (unsigned)ldin, etot);

    // GEMM1: t1 = (h + agg) @ w1 + b1   (+ stats for BN1)
    k_gin_gemm<<<gemmBlocks, 128, 0, stream>>>(
        hin, agg, ldin, ldin, KP, w1, Kw1, b1,
        nullptr, nullptr, nullptr, nullptr, /*useBN=*/0, invN,
        t1, stats + 0, stats + 64, N);

    // GEMM2: t2 = relu(bn1(t1)) @ w2 + b2   (+ stats for BN2)
    k_gin_gemm<<<gemmBlocks, 128, 0, stream>>>(
        t1, nullptr, D_H, D_H, D_H, w2, D_H, b2,
        stats + 0, stats + 64, gl + 0, bl + 0, /*useBN=*/1, invN,
        t2, stats + 128, stats + 192, N);

    // BN2 + ReLU (in place), stats for BN3
    k_bn_relu<<<bnBlocks, 256, 0, stream>>>(
        t2, stats + 128, stats + 192, gl + 64, bl + 64, invN,
        t2, stats + 256, stats + 320, nullptr, nullptr, N);

    // BN3 + ReLU -> next h, fused graph pooling
    k_bn_relu<<<bnBlocks, 256, 0, stream>>>(
        t2, stats + 256, stats + 320, gl + 128, bl + 128, invN,
        hc, nullptr, nullptr, pooledR + (size_t)l * G * D_H, batch, N);
  }

  // heads + L2 normalize
  k_score<<<cdiv_i(G, 128), 128, 0, stream>>>(pooled0, pooledR, pw0, pb0, pwr, pbr,
                                              (float*)d_out, G);
  (void)n_in; (void)out_size; (void)ws_size;
}
